// WindowAttentionBlock_24094766530838
// MI455X (gfx1250) — compile-verified, matrix operand order checked by reading, not run on hardware
//
#include <hip/hip_runtime.h>
#include <hip/hip_bf16.h>

typedef __attribute__((ext_vector_type(16))) _Float16 v16h;
typedef __attribute__((ext_vector_type(8)))  _Float16 v8h;
typedef __attribute__((ext_vector_type(8)))  float    v8f;
typedef __attribute__((ext_vector_type(4)))  unsigned v4u;
typedef __attribute__((ext_vector_type(8)))  int      v8i;
typedef __attribute__((ext_vector_type(4)))  int      v4i;

// D = A(16x32 f16) * B(32x16 f16) + C(16x16 f32)
__device__ __forceinline__ v8f wmma_f16(v16h a, v16h b, v8f c) {
  return __builtin_amdgcn_wmma_f32_16x16x32_f16(false, a, false, b, (short)0, c,
                                                false, false);
}

#if __has_builtin(__builtin_amdgcn_tensor_load_to_lds)
#define HAVE_TDM 1
#else
#define HAVE_TDM 0
#endif

#if HAVE_TDM
// One 2D TDM descriptor: load a (64 x K)-f16 row-major panel at gptr into LDS.
// D# layout per CDNA5 ISA.
__device__ __forceinline__ void tdm_load_panel64xK(const _Float16* gptr,
                                                   unsigned lds_off, int K) {
  unsigned long long ga = (unsigned long long)gptr;
  v4u g0;
  g0[0] = 1u;                                    // count=1, no gather
  g0[1] = lds_off;                               // lds_addr (bytes)
  g0[2] = (unsigned)ga;                          // global_addr[31:0]
  g0[3] = (unsigned)((ga >> 32) & 0x01FFFFFFull) // global_addr[56:32]
          | (2u << 30);                          // type=2 ("image")
  v8i g1;
  g1[0] = 0x10000;                               // data_size=1 -> 2 bytes
  g1[1] = (K & 0xFFFF) << 16;                    // tensor_dim0[15:0]
  g1[2] = ((unsigned)K >> 16) | (64u << 16);     // dim0[31:16] | dim1[15:0]
  g1[3] = (K & 0xFFFF) << 16;                    // dim1[31:16]=0 | tile_dim0=K
  g1[4] = 64;                                    // tile_dim1=64, tile_dim2=0
  g1[5] = K;                                     // tensor_dim0_stride[31:0]
  g1[6] = 0;                                     // stride hi | dim1_stride lo
  g1[7] = 0;
  v4i g2 = {0, 0, 0, 0};                         // 2D: groups 2/3 unused
  v4i g3 = {0, 0, 0, 0};
#if defined(__clang_major__) && __clang_major__ >= 23
  v8i g4 = {0, 0, 0, 0, 0, 0, 0, 0};
  __builtin_amdgcn_tensor_load_to_lds(g0, g1, g2, g3, g4, 0);
#else
  __builtin_amdgcn_tensor_load_to_lds(g0, g1, g2, g3, 0);
#endif
}
__device__ __forceinline__ void wait_tensorcnt0() {
#if __has_builtin(__builtin_amdgcn_s_wait_tensorcnt)
  __builtin_amdgcn_s_wait_tensorcnt(0);
#else
  asm volatile("s_wait_tensorcnt 0x0" ::: "memory");
#endif
}
#endif

// ---------------------------------------------------------------------------
// Constants for this problem size
// ---------------------------------------------------------------------------
#define BATCH   16
#define INC     128
#define C_      256
#define HEADS   8
#define DHEAD   32
#define H_      56
#define W_      56
#define NPIX    (H_ * W_)          // 3136
#define HP      58                 // padded spatial
#define WS_     7
#define NWIN    (BATCH * 8 * 8)    // 1024 windows
#define NTOK    49
#define NPAD    64                 // padded window tokens
#define HID     1024
#define QKVN    (3 * C_)           // 768
#define MTOK    (BATCH * NPIX)     // 50176 tokens (multiple of 128)

// ===========================================================================
// 0a. pad + cast x: NCHW f32 -> padded NHWC f16 (16,58,58,128), zero border
// ===========================================================================
__global__ __launch_bounds__(256) void k_pad_cast_x(const float* __restrict__ x,
                                                    _Float16* __restrict__ xp) {
  unsigned i = blockIdx.x * 256u + threadIdx.x;      // over 16*58*58*128
  if (i >= (unsigned)(BATCH * HP * HP * INC)) return;
  int c  = i & (INC - 1);
  int wp = (i >> 7) % HP;
  int hp = (i / (INC * HP)) % HP;
  int b  = i / (INC * HP * HP);
  float v = 0.f;
  if (hp >= 1 && hp <= H_ && wp >= 1 && wp <= W_)
    v = x[(((size_t)b * INC + c) * H_ + (hp - 1)) * W_ + (wp - 1)];
  xp[i] = (_Float16)v;
}

// ===========================================================================
// 0b. conv weights (256,128,3,3) f32 -> wc[tap][oc][ic] f16
// ===========================================================================
__global__ __launch_bounds__(256) void k_conv_w(const float* __restrict__ w,
                                                _Float16* __restrict__ wc) {
  unsigned i = blockIdx.x * 256u + threadIdx.x;      // 9*256*128
  if (i >= 9u * C_ * INC) return;
  int ic  = i & (INC - 1);
  int oc  = (i >> 7) & (C_ - 1);
  int tap = i >> 15;
  wc[i] = (_Float16)w[((size_t)oc * INC + ic) * 9 + tap];
}

// ===========================================================================
// 0c. generic f32 -> f16 cast (weights)
// ===========================================================================
__global__ __launch_bounds__(256) void k_cast(const float* __restrict__ s,
                                              _Float16* __restrict__ d, int n) {
  int i = blockIdx.x * 256 + threadIdx.x;
  if (i < n) d[i] = (_Float16)s[i];
}

// ===========================================================================
// 1. conv3x3 implicit GEMM (WMMA) + bias + LN1 + window partition
//    block = 256 threads (8 waves). Tile: M=32 pixels x N=256 channels.
//    grid.x = BATCH * (NPIX/32). Weights (0.6MB) stay L2-resident.
// ===========================================================================
__global__ __launch_bounds__(256) void k_conv_ln1(const _Float16* __restrict__ xp,
                                                  const _Float16* __restrict__ wc,
                                                  const float* __restrict__ bias,
                                                  const float* __restrict__ g1,
                                                  const float* __restrict__ b1,
                                                  _Float16* __restrict__ xw) {
  extern __shared__ char smem_raw[];
  float* tile = (float*)smem_raw;                    // 32 x 256 f32 = 32 KB
  const int lane = threadIdx.x & 31, wv = threadIdx.x >> 5;
  const int lm = lane & 15, hh = lane >> 4;
  const int b = blockIdx.x / (NPIX / 32);
  const int pixel_base = (blockIdx.x % (NPIX / 32)) * 32;
  const int m_sub = wv & 1;                          // which 16-pixel half
  const int n_base = (wv >> 1) * 64;                 // 64 output channels

  const int p  = pixel_base + m_sub * 16 + lm;       // this lane's A-row pixel
  const int ph = p / W_, pw = p % W_;

  v8f acc[4] = {};
#pragma unroll
  for (int ky = 0; ky < 3; ++ky) {
#pragma unroll
    for (int kx = 0; kx < 3; ++kx) {
      const _Float16* abase =
          xp + ((size_t)((b * HP + ph + ky) * HP) + (pw + kx)) * INC + hh * 16;
      const _Float16* bbase =
          wc + ((size_t)((ky * 3 + kx) * C_ + n_base + lm)) * INC + hh * 16;
#pragma unroll
      for (int kc = 0; kc < 4; ++kc) {               // K = 128 = 4 x 32
        v16h av = *(const v16h*)(abase + kc * 32);
#pragma unroll
        for (int t = 0; t < 4; ++t) {
          v16h bv = *(const v16h*)(bbase + (size_t)t * 16 * INC + kc * 32);
          acc[t] = wmma_f16(av, bv, acc[t]);
        }
      }
    }
  }
  // bias + stash tile to LDS (f32)
#pragma unroll
  for (int t = 0; t < 4; ++t) {
    int col = n_base + t * 16 + lm;
    float bb = bias[col];
#pragma unroll
    for (int r = 0; r < 8; ++r) {
      int row = m_sub * 16 + r + 8 * hh;
      tile[row * C_ + col] = acc[t][r] + bb;
    }
  }
  __syncthreads();

  // LayerNorm over 256 channels: one wave per row (8 waves x 4 rows)
#pragma unroll
  for (int rr = 0; rr < 4; ++rr) {
    int row = wv * 4 + rr;
    int c0 = lane * 8;
    float v[8];
    float s = 0.f, sq = 0.f;
#pragma unroll
    for (int u = 0; u < 8; ++u) {
      v[u] = tile[row * C_ + c0 + u];
      s += v[u]; sq += v[u] * v[u];
    }
#pragma unroll
    for (int m = 16; m >= 1; m >>= 1) {
      s  += __shfl_xor(s, m, 32);
      sq += __shfl_xor(sq, m, 32);
    }
    float mu = s * (1.f / C_);
    float var = sq * (1.f / C_) - mu * mu;
    float rstd = rsqrtf(var + 1e-5f);
    int pp = pixel_base + row;
    int h = pp / W_, w = pp % W_;
    int win = b * 64 + (h / WS_) * 8 + (w / WS_);
    int trow = (h % WS_) * WS_ + (w % WS_);
    v8h o;
#pragma unroll
    for (int u = 0; u < 8; ++u)
      o[u] = (_Float16)((v[u] - mu) * rstd * g1[c0 + u] + b1[c0 + u]);
    *(v8h*)(xw + ((size_t)(win * NPAD + trow)) * C_ + c0) = o;
  }
}

// ===========================================================================
// 2. generic GEMM: C(MxN) = A(MxK) @ B(NxK)^T [+bias] [gelu], f16 in/out.
//    Block tile: M=128 x N=64 (8 waves, one 16x64 tile each). The 64xK f16
//    weight panel is staged in LDS once per block via the Tensor Data Mover
//    (tensor_load_to_lds + s_wait_tensorcnt), so the K-loop feeds WMMA from
//    ds_load_b128 with only A-traffic on the VMEM path.
//    grid = (M/128)*(N/64), dyn LDS = K*128 bytes.
// ===========================================================================
__global__ __launch_bounds__(256) void k_gemm(const _Float16* __restrict__ A,
                                              const _Float16* __restrict__ B,
                                              _Float16* __restrict__ C,
                                              const float* __restrict__ bias,
                                              int M, int N, int K, int act) {
  extern __shared__ char smem_raw[];
  _Float16* Bl = (_Float16*)smem_raw;            // 64 x K f16 panel
  const int lane = threadIdx.x & 31, wv = threadIdx.x >> 5;
  const int lm = lane & 15, hh = lane >> 4;
  const int nt64 = N >> 6;
  const int mb  = blockIdx.x / nt64;             // 128-row block
  const int n64 = blockIdx.x % nt64;             // 64-col block
  const _Float16* Bpanel = B + (size_t)n64 * 64 * K;

#if HAVE_TDM
  if (wv == 0) {                                 // one TDM op per workgroup
    tdm_load_panel64xK(Bpanel, (unsigned)(size_t)smem_raw, K);
    wait_tensorcnt0();
  }
#else
  for (int idx = threadIdx.x; idx < K * 8; idx += 256)
    *(v8h*)(Bl + idx * 8) = *(const v8h*)(Bpanel + idx * 8);
#endif
  __syncthreads();

  const int mt16 = mb * 8 + wv;                  // this wave's 16-row tile
  const _Float16* arow = A + (size_t)(mt16 * 16 + lm) * K + hh * 16;
  v8f acc[4] = {};
  for (int k0 = 0; k0 < K; k0 += 32) {
    v16h av = *(const v16h*)(arow + k0);
#pragma unroll
    for (int t = 0; t < 4; ++t) {
      v16h bv = *(const v16h*)(Bl + (size_t)(t * 16 + lm) * K + k0 + hh * 16);
      acc[t] = wmma_f16(av, bv, acc[t]);
    }
  }
#pragma unroll
  for (int t = 0; t < 4; ++t) {
    int col = n64 * 64 + t * 16 + lm;
    float bb = bias ? bias[col] : 0.f;
#pragma unroll
    for (int r = 0; r < 8; ++r) {
      int row = mt16 * 16 + r + 8 * hh;
      float v = acc[t][r] + bb;
      if (act) v = 0.5f * v * (1.f + erff(v * 0.70710678118f));  // exact GELU
      C[(size_t)row * N + col] = (_Float16)v;
    }
  }
}

// ===========================================================================
// 3. transpose V part of qkv -> vT[win][head][dim(32)][tok(64)] f16
// ===========================================================================
__global__ __launch_bounds__(256) void k_vt(const _Float16* __restrict__ qkv,
                                            _Float16* __restrict__ vT) {
  unsigned i = blockIdx.x * 256u + threadIdx.x;  // 1024*8*32*64 = 16777216
  int k = i & 63;
  int n = (i >> 6) & 31;
  int h = (i >> 11) & 7;
  int w = i >> 14;
  vT[i] = qkv[((size_t)w * NPAD + k) * QKVN + 2 * C_ + h * DHEAD + n];
}

// ===========================================================================
// 4. window attention: block = 1 window, 8 waves = 8 heads
//    logits (64x64) via WMMA (K=32 single shot), softmax in regs+shfl,
//    probs -> LDS f16, P@V via WMMA from LDS.
// ===========================================================================
__global__ __launch_bounds__(256) void k_attn(const _Float16* __restrict__ qkv,
                                              const _Float16* __restrict__ vT,
                                              const float* __restrict__ relb,
                                              _Float16* __restrict__ attn_o) {
  extern __shared__ char smem_raw[];
  _Float16* Pall = (_Float16*)smem_raw;                    // 8 * 4096 f16
  float* tbl = (float*)(smem_raw + 8 * NPAD * NPAD * 2);   // 1352 f32
  const int lane = threadIdx.x & 31, wv = threadIdx.x >> 5;
  const int lm = lane & 15, hh = lane >> 4;
  const int win = blockIdx.x;

  for (int t = threadIdx.x; t < 169 * HEADS; t += 256) tbl[t] = relb[t];
  __syncthreads();

  const _Float16* qb = qkv + (size_t)win * NPAD * QKVN + wv * DHEAD;
  const _Float16* kb = qb + C_;

  v8f acc[4][4] = {};
#pragma unroll
  for (int i = 0; i < 4; ++i) {
    v16h av = *(const v16h*)(qb + (size_t)(i * 16 + lm) * QKVN + hh * 16);
#pragma unroll
    for (int j = 0; j < 4; ++j) {
      v16h bv = *(const v16h*)(kb + (size_t)(j * 16 + lm) * QKVN + hh * 16);
      acc[i][j] = wmma_f16(av, bv, acc[i][j]);
    }
  }

  const float scale = 0.17677669529663687f;  // 1/sqrt(32)
  _Float16* Pw = Pall + wv * NPAD * NPAD;
#pragma unroll
  for (int i = 0; i < 4; ++i) {
#pragma unroll
    for (int r = 0; r < 8; ++r) {
      int m = i * 16 + r + 8 * hh;
      int my = m / WS_, mx = m - my * WS_;
      bool mval = m < NTOK;
      float lv[4];
#pragma unroll
      for (int j = 0; j < 4; ++j) {
        int n = j * 16 + lm;
        float l = acc[i][j][r] * scale;
        if (n < NTOK) {
          if (mval) {
            int ny = n / WS_, nx = n - ny * WS_;
            l += tbl[((my - ny + 6) * 13 + (mx - nx + 6)) * HEADS + wv];
          }
        } else {
          l = -1e30f;  // masked padded column
        }
        lv[j] = l;
      }
      float mx4 = fmaxf(fmaxf(lv[0], lv[1]), fmaxf(lv[2], lv[3]));
#pragma unroll
      for (int s = 8; s >= 1; s >>= 1) mx4 = fmaxf(mx4, __shfl_xor(mx4, s, 32));
      float e[4], sum = 0.f;
#pragma unroll
      for (int j = 0; j < 4; ++j) { e[j] = __expf(lv[j] - mx4); sum += e[j]; }
#pragma unroll
      for (int s = 8; s >= 1; s >>= 1) sum += __shfl_xor(sum, s, 32);
      float inv = 1.f / sum;
#pragma unroll
      for (int j = 0; j < 4; ++j)
        Pw[m * NPAD + j * 16 + lm] = (_Float16)(e[j] * inv);
    }
  }
  __syncthreads();

  // O(64x32) = P(64x64) @ V(64x32); B from vT (dim-major, contiguous tokens)
  const _Float16* vb = vT + ((size_t)win * HEADS + wv) * DHEAD * NPAD;
  _Float16* ob = attn_o + (size_t)win * NPAD * C_ + wv * DHEAD;
#pragma unroll
  for (int i = 0; i < 4; ++i) {
    v8f o0 = {}, o1 = {};
#pragma unroll
    for (int kc = 0; kc < 2; ++kc) {
      v16h av = *(const v16h*)(Pw + (i * 16 + lm) * NPAD + kc * 32 + hh * 16);
      v16h b0 = *(const v16h*)(vb + (size_t)lm * NPAD + kc * 32 + hh * 16);
      v16h b1 = *(const v16h*)(vb + (size_t)(16 + lm) * NPAD + kc * 32 + hh * 16);
      o0 = wmma_f16(av, b0, o0);
      o1 = wmma_f16(av, b1, o1);
    }
#pragma unroll
    for (int r = 0; r < 8; ++r) {
      int m = i * 16 + r + 8 * hh;
      ob[(size_t)m * C_ + lm]      = (_Float16)o0[r];
      ob[(size_t)m * C_ + 16 + lm] = (_Float16)o1[r];
    }
  }
}

// ===========================================================================
// 5. window reverse + LN2: windowed (nW*64,256) f16 -> token-major f16
// ===========================================================================
__global__ __launch_bounds__(256) void k_ln2(const _Float16* __restrict__ src,
                                             const float* __restrict__ g,
                                             const float* __restrict__ b,
                                             _Float16* __restrict__ dst) {
  const int lane = threadIdx.x & 31, wv = threadIdx.x >> 5;
  int t = blockIdx.x * 8 + wv;
  int bi = t / NPIX, p = t % NPIX;
  int h = p / W_, w = p % W_;
  int win = bi * 64 + (h / WS_) * 8 + (w / WS_);
  int trow = (h % WS_) * WS_ + (w % WS_);
  const _Float16* s = src + ((size_t)win * NPAD + trow) * C_ + lane * 8;
  float v[8], su = 0.f, sq = 0.f;
  v8h in = *(const v8h*)s;
#pragma unroll
  for (int u = 0; u < 8; ++u) { v[u] = (float)in[u]; su += v[u]; sq += v[u]*v[u]; }
#pragma unroll
  for (int m = 16; m >= 1; m >>= 1) { su += __shfl_xor(su, m, 32); sq += __shfl_xor(sq, m, 32); }
  float mu = su * (1.f / C_);
  float rstd = rsqrtf(sq * (1.f / C_) - mu * mu + 1e-5f);
  int c0 = lane * 8;
  v8h o;
#pragma unroll
  for (int u = 0; u < 8; ++u)
    o[u] = (_Float16)((v[u] - mu) * rstd * g[c0 + u] + b[c0 + u]);
  *(v8h*)(dst + (size_t)t * C_ + c0) = o;
}

// ===========================================================================
// 6. LN3 + transpose to NCHW f32 output (x = x + x is a no-op under LN)
// ===========================================================================
__global__ __launch_bounds__(256) void k_ln3_out(const _Float16* __restrict__ src,
                                                 const float* __restrict__ g,
                                                 const float* __restrict__ b,
                                                 float* __restrict__ out) {
  const int lane = threadIdx.x & 31, wv = threadIdx.x >> 5;
  int t = blockIdx.x * 8 + wv;
  const _Float16* s = src + (size_t)t * C_ + lane * 8;
  float v[8], su = 0.f, sq = 0.f;
  v8h in = *(const v8h*)s;
#pragma unroll
  for (int u = 0; u < 8; ++u) { v[u] = (float)in[u]; su += v[u]; sq += v[u]*v[u]; }
#pragma unroll
  for (int m = 16; m >= 1; m >>= 1) { su += __shfl_xor(su, m, 32); sq += __shfl_xor(sq, m, 32); }
  float mu = su * (1.f / C_);
  float rstd = rsqrtf(sq * (1.f / C_) - mu * mu + 1e-5f);
  int bi = t / NPIX, p = t % NPIX;
  int h = p / W_, w = p % W_;
  int c0 = lane * 8;
#pragma unroll
  for (int u = 0; u < 8; ++u) {
    float y = (v[u] - mu) * rstd * g[c0 + u] + b[c0 + u];
    out[(((size_t)bi * C_ + c0 + u) * H_ + h) * W_ + w] = y;
  }
}

// ===========================================================================
// launch
// ===========================================================================
extern "C" void kernel_launch(void* const* d_in, const int* in_sizes, int n_in,
                              void* d_out, int out_size, void* d_ws, size_t ws_size,
                              hipStream_t stream) {
  (void)in_sizes; (void)n_in; (void)out_size; (void)ws_size;
  const float* x      = (const float*)d_in[0];
  const float* conv_w = (const float*)d_in[1];
  const float* conv_b = (const float*)d_in[2];
  const float* g1 = (const float*)d_in[3];
  const float* b1 = (const float*)d_in[4];
  const float* g2 = (const float*)d_in[5];
  const float* b2 = (const float*)d_in[6];
  const float* g3 = (const float*)d_in[7];
  const float* b3 = (const float*)d_in[8];
  const float* qkv_w  = (const float*)d_in[9];
  const float* relb   = (const float*)d_in[10];
  const float* proj_w = (const float*)d_in[11];
  const float* proj_b = (const float*)d_in[12];
  const float* fc1_w  = (const float*)d_in[13];
  const float* fc1_b  = (const float*)d_in[14];
  const float* fc2_w  = (const float*)d_in[15];
  const float* fc2_b  = (const float*)d_in[16];
  float* out = (float*)d_out;

  // workspace bump allocator (256B aligned)
  char* ws = (char*)d_ws;
  auto alloc = [&](size_t bytes) {
    char* p = ws; ws += (bytes + 255) & ~(size_t)255; return p;
  };
  _Float16* xp  = (_Float16*)alloc((size_t)BATCH * HP * HP * INC * 2);   // 13.8 MB
  _Float16* wc  = (_Float16*)alloc((size_t)9 * C_ * INC * 2);
  _Float16* wq  = (_Float16*)alloc((size_t)QKVN * C_ * 2);
  _Float16* wpj = (_Float16*)alloc((size_t)C_ * C_ * 2);
  _Float16* w1  = (_Float16*)alloc((size_t)HID * C_ * 2);
  _Float16* w2  = (_Float16*)alloc((size_t)C_ * HID * 2);
  const size_t szR1 = (size_t)NWIN * NPAD * C_ * 2;        // 33.5 MB
  _Float16* R1 = (_Float16*)alloc(szR1);                   // xw -> attn_o -> out2
  _Float16* R2 = (_Float16*)alloc((size_t)MTOK * HID * 2); // qkv -> proj_o -> h1
  _Float16* R3 = (_Float16*)alloc(szR1);                   // vT -> ln2

  // --- casts ---
  k_pad_cast_x<<<(BATCH * HP * HP * INC + 255) / 256, 256, 0, stream>>>(x, xp);
  k_conv_w<<<(9 * C_ * INC + 255) / 256, 256, 0, stream>>>(conv_w, wc);
  k_cast<<<(QKVN * C_ + 255) / 256, 256, 0, stream>>>(qkv_w, wq, QKVN * C_);
  k_cast<<<(C_ * C_ + 255) / 256, 256, 0, stream>>>(proj_w, wpj, C_ * C_);
  k_cast<<<(HID * C_ + 255) / 256, 256, 0, stream>>>(fc1_w, w1, HID * C_);
  k_cast<<<(C_ * HID + 255) / 256, 256, 0, stream>>>(fc2_w, w2, C_ * HID);

  // zero xw so padded window rows 49..63 are 0
  hipMemsetAsync(R1, 0, szR1, stream);

  // --- conv + LN1 + window partition ---
  k_conv_ln1<<<BATCH * (NPIX / 32), 256, 32 * C_ * 4, stream>>>(
      xp, wc, conv_b, g1, b1, R1);

  // --- QKV GEMM: (nW*64,256) x (768,256)^T -> R2 ---
  k_gemm<<<(NWIN * NPAD / 128) * (QKVN / 64), 256, C_ * 128, stream>>>(
      R1, wq, R2, nullptr, NWIN * NPAD, QKVN, C_, 0);

  // --- V transpose ---
  k_vt<<<(NWIN * HEADS * DHEAD * NPAD) / 256, 256, 0, stream>>>(R2, R3);

  // --- attention (writes attn_o into R1, overwriting dead xw) ---
  size_t attn_lds = (size_t)HEADS * NPAD * NPAD * 2 + 169 * HEADS * 4;
  k_attn<<<NWIN, 256, attn_lds, stream>>>(R2, R3, relb, R1);

  // --- proj GEMM (windowed) -> R2 (qkv dead) ---
  k_gemm<<<(NWIN * NPAD / 128) * (C_ / 64), 256, C_ * 128, stream>>>(
      R1, wpj, R2, proj_b, NWIN * NPAD, C_, C_, 0);

  // --- window reverse + LN2 -> R3 (vT dead) ---
  k_ln2<<<MTOK / 8, 256, 0, stream>>>(R2, g2, b2, R3);

  // --- FC1 + exact GELU -> R2 (proj_o dead) ---
  k_gemm<<<(MTOK / 128) * (HID / 64), 256, C_ * 128, stream>>>(
      R3, w1, R2, fc1_b, MTOK, HID, C_, 1);

  // --- FC2 -> R1 (attn_o dead) ---
  k_gemm<<<(MTOK / 128) * (C_ / 64), 256, HID * 128, stream>>>(
      R2, w2, R1, fc2_b, MTOK, C_, HID, 0);

  // --- LN3 + NCHW f32 output (x+x no-op under LN) ---
  k_ln3_out<<<MTOK / 8, 256, 0, stream>>>(R1, g3, b3, out);
}
